// GCNPCNET_39058432590327
// MI455X (gfx1250) — compile-verified, matrix operand order checked by reading, not run on hardware
//
#include <hip/hip_runtime.h>
#include <hip/hip_bf16.h>
#include <cstdint>

typedef __attribute__((ext_vector_type(16))) _Float16 v16h;
typedef __attribute__((ext_vector_type(8)))  float    v8f;

static constexpr int BB   = 16;    // batch
static constexpr int NN   = 1024;  // points
static constexpr int KNBR = 20;    // neighbors
static constexpr float SLOPE = 0.2f;
static constexpr float EPS   = 1e-5f;

// ---------------------------------------------------------------------------
// Optional CDNA5 Tensor Data Mover path (compile-safe probes)
// ---------------------------------------------------------------------------
#if __has_builtin(__builtin_amdgcn_tensor_load_to_lds) && \
    __has_builtin(__builtin_amdgcn_s_wait_tensorcnt)
#define USE_TDM 1
#if __has_include(<hip/amd_detail/amd_gfx1250_TDM.h>)
#define TDM_SIX_ARGS 1
#else
#define TDM_SIX_ARGS 0
#endif
#else
#define USE_TDM 0
#endif

#if USE_TDM
typedef unsigned int u32x4 __attribute__((ext_vector_type(4)));
typedef int          i32x4 __attribute__((ext_vector_type(4)));
typedef int          i32x8 __attribute__((ext_vector_type(8)));

// Stage `elems` contiguous f16 values from global to LDS via TDM.
// D# per CDNA5 ISA ch.8: 1-line 2D tile, data_size=2B, no padding/multicast.
__device__ __forceinline__ void tdm_load_to_lds(const _Float16* gsrc,
                                                unsigned lds_byte_addr,
                                                int elems) {
    unsigned long long ga = (unsigned long long)(uintptr_t)gsrc;
    u32x4 g0;
    g0.x = 1u;                                            // count=1 (user mode)
    g0.y = lds_byte_addr;                                 // lds_addr
    g0.z = (unsigned)(ga & 0xFFFFFFFFu);                  // global_addr[31:0]
    g0.w = (unsigned)((ga >> 32) & 0x1FFFFFFu) | (2u << 30); // addr[56:32] | type=2
    i32x8 g1;
    g1[0] = (1 << 16);                      // wg_mask=0, data_size=1 (2 bytes)
    g1[1] = (elems & 0xFFFF) << 16;         // tensor_dim0[15:0] @ bits63:48
    g1[2] = ((elems >> 16) & 0xFFFF) | (1 << 16); // tensor_dim0[31:16], tensor_dim1=1
    g1[3] = (elems & 0xFFFF) << 16;         // tile_dim0 = elems (<=8192)
    g1[4] = 1;                              // tile_dim1=1, tile_dim2=0
    g1[5] = elems;                          // tensor_dim0_stride[31:0]
    g1[6] = 0;
    g1[7] = 0;
    i32x4 z4 = {0, 0, 0, 0};
#if TDM_SIX_ARGS
    i32x8 z8 = {0, 0, 0, 0, 0, 0, 0, 0};
    __builtin_amdgcn_tensor_load_to_lds(g0, g1, z4, z4, z8, 0);
#else
    __builtin_amdgcn_tensor_load_to_lds(g0, g1, z4, z4, 0);
#endif
}
#endif  // USE_TDM

// ---------------------------------------------------------------------------
// WMMA fragment loader: 16-bit A/B 16x16x32 lane layout (CDNA5 ISA 7.12.2).
// lane<16  -> row = lane,    K = {k0..k0+7, k0+16..k0+23}
// lane>=16 -> row = lane-16, K = {k0+8..k0+15, k0+24..k0+31}
// ---------------------------------------------------------------------------
__device__ __forceinline__ v16h load_frag16(const _Float16* rowp, int k0, int hi) {
    const _Float16* p = rowp + k0 + (hi ? 8 : 0);
    v16h r;
#pragma unroll
    for (int i = 0; i < 8; ++i) r[i] = p[i];
#pragma unroll
    for (int i = 0; i < 8; ++i) r[i + 8] = p[i + 16];
    return r;
}

__device__ __forceinline__ v8f wmma16(v16h a, v16h b, v8f c) {
    return __builtin_amdgcn_wmma_f32_16x16x32_f16(false, a, false, b, (short)0, c,
                                                  false, false);
}

// ---------------------------------------------------------------------------
// Layout/cast kernels
// ---------------------------------------------------------------------------
__global__ void cast_bcn_to_bnc_f16(const float* __restrict__ src,
                                    _Float16* __restrict__ dst,
                                    int B, int C, int N, int Cpad, int srcCtot) {
    long i = (long)blockIdx.x * blockDim.x + threadIdx.x;
    long total = (long)B * N * Cpad;
    if (i >= total) return;
    int c = (int)(i % Cpad);
    long t = i / Cpad;
    int n = (int)(t % N);
    int b = (int)(t / N);
    float v = (c < C) ? src[((long)b * srcCtot + c) * N + n] : 0.0f;
    dst[i] = (_Float16)v;
}

__global__ void cast_mat_f16(const float* __restrict__ W,
                             _Float16* __restrict__ dst,
                             int O, int C, int Opad, int Cpad) {
    long i = (long)blockIdx.x * blockDim.x + threadIdx.x;
    long total = (long)Opad * Cpad;
    if (i >= total) return;
    int c = (int)(i % Cpad);
    int o = (int)(i / Cpad);
    float v = (o < O && c < C) ? W[(long)o * C + c] : 0.0f;
    dst[i] = (_Float16)v;
}

__global__ void sqnorm_kernel(const float* __restrict__ src, float* __restrict__ sq,
                              int B, int C, int N, int srcCtot) {
    int i = blockIdx.x * blockDim.x + threadIdx.x;
    if (i >= B * N) return;
    int b = i / N, n = i % N;
    float s = 0.0f;
    for (int c = 0; c < C; ++c) {
        float v = src[((long)b * srcCtot + c) * N + n];
        s += v * v;
    }
    sq[i] = s;
}

// ---------------------------------------------------------------------------
// Fused Gram (WMMA) + neg-distance + per-row top-20. A fragments held in
// registers across the whole column sweep (CP <= 128 -> <= 32 VGPRs).
// ---------------------------------------------------------------------------
template <int CP>
__global__ void knn_topk_kernel(const _Float16* __restrict__ xh,  // (B,N,CP)
                                const float* __restrict__ sq,     // (B,N)
                                int* __restrict__ nbr,            // (B,N,KNBR)
                                int N) {
    constexpr int KT = CP / 32;
    int rowTiles = N / 16;
    int b  = blockIdx.x / rowTiles;
    int m0 = (blockIdx.x % rowTiles) * 16;
    int lane = threadIdx.x;
    int hi = lane >> 4;
    int lr = lane & 15;

    const _Float16* xb = xh + (long)b * N * CP;
    const float*    sb = sq + (long)b * N;

    v16h afr[KT];
#pragma unroll
    for (int k = 0; k < KT; ++k)
        afr[k] = load_frag16(xb + (long)(m0 + lr) * CP, k * 32, hi);

    float best[KNBR];
    int   bidx[KNBR];
#pragma unroll
    for (int t = 0; t < KNBR; ++t) { best[t] = -3.402823466e38f; bidx[t] = 0; }

    float rsq = sb[m0 + lr];

    __shared__ float tile[16][17];

    for (int n0 = 0; n0 < N; n0 += 16) {
        v16h bfr[KT];
#pragma unroll
        for (int k = 0; k < KT; ++k)
            bfr[k] = load_frag16(xb + (long)(n0 + lr) * CP, k * 32, hi);
        v8f acc = {};
#pragma unroll
        for (int k = 0; k < KT; ++k) acc = wmma16(afr[k], bfr[k], acc);
#pragma unroll
        for (int r = 0; r < 8; ++r) tile[hi * 8 + r][lr] = acc[r];
        __syncthreads();

        if (lane < 16) {
            for (int j = 0; j < 16; ++j) {
                int col = n0 + j;
                float v = 2.0f * tile[lane][j] - rsq - sb[col];
                if (v > best[KNBR - 1]) {
                    best[KNBR - 1] = v;
                    bidx[KNBR - 1] = col;
#pragma unroll
                    for (int t = KNBR - 1; t > 0; --t) {
                        if (best[t] > best[t - 1]) {
                            float tv = best[t]; best[t] = best[t - 1]; best[t - 1] = tv;
                            int   ti = bidx[t]; bidx[t] = bidx[t - 1]; bidx[t - 1] = ti;
                        }
                    }
                }
            }
        }
        __syncthreads();
    }

    if (lane < 16) {
        int* out = nbr + ((long)b * N + m0 + lane) * KNBR;
#pragma unroll
        for (int t = 0; t < KNBR; ++t) out[t] = bidx[t];
    }
}

// agg[b,c,n] = sum_j src[b,c,nbr[b,n,j]]
__global__ void gather_sum_kernel(const float* __restrict__ src,
                                  const int* __restrict__ nbr,
                                  float* __restrict__ agg,
                                  int B, int C, int N, int srcCtot) {
    int bn = blockIdx.x;
    int b = bn / N, n = bn % N;
    const int* id = nbr + ((long)b * N + n) * KNBR;
    int ids[KNBR];
#pragma unroll
    for (int j = 0; j < KNBR; ++j) ids[j] = id[j];
    for (int c = threadIdx.x; c < C; c += blockDim.x) {
        const float* row = src + ((long)b * srcCtot + c) * N;
        float s = 0.0f;
#pragma unroll
        for (int j = 0; j < KNBR; ++j) s += row[ids[j]];
        agg[((long)b * C + c) * N + n] = s;
    }
}

// ---------------------------------------------------------------------------
// GEMM: y[b,o,n] = sum_c Wh[o,c]*Xh[b,n,c]. 4 waves/block; weights staged to
// LDS (TDM when available); each wave computes 4 N-tiles per A fragment.
// Templated on CP so the K loop fully unrolls (batched loads, compile-time
// prefetch, exact LDS footprint). BN stats via LDS atomics -> global atomics.
// ---------------------------------------------------------------------------
template <int CP>
__global__ void gemm_bn_kernel(const _Float16* __restrict__ Wh,   // (O, CP)
                               const _Float16* __restrict__ Xh,   // (B, N, CP)
                               float* __restrict__ y,             // (B, O, N)
                               float* __restrict__ chsum, float* __restrict__ chsq,
                               int B, int O, int N) {
    constexpr int KT = CP / 32;
    __shared__ _Float16 wtile[16 * CP];
    __shared__ float lsum[16];
    __shared__ float lss[16];

    int nBlocks = N / 256, tilesO = O / 16;
    int t = blockIdx.x;
    int nb = t % nBlocks;  t /= nBlocks;
    int o0 = (t % tilesO) * 16;
    int b  = t / tilesO;

    int tid  = threadIdx.x;
    int lane = tid & 31;
    int wid  = tid >> 5;
    int hi = lane >> 4, lr = lane & 15;

    const _Float16* wrow = Wh + (long)o0 * CP;
    constexpr int elems = 16 * CP;

    if (tid < 16) { lsum[tid] = 0.0f; lss[tid] = 0.0f; }

#if USE_TDM
    if (tid < 32) {  // wave 0 issues the TDM descriptor; others branch over
        tdm_load_to_lds(wrow, (unsigned)(uintptr_t)(void*)wtile, elems);
        __builtin_amdgcn_s_wait_tensorcnt(0);
    }
#else
    for (int i = tid * 8; i < elems; i += blockDim.x * 8) {
        *(float4*)(wtile + i) = *(const float4*)(wrow + i);
    }
#endif
    __syncthreads();

    const _Float16* xb = Xh + (long)b * N * CP;
    int nBase = nb * 256 + wid * 64;
    const _Float16* br0 = xb + (long)(nBase +  0 + lr) * CP;
    const _Float16* br1 = xb + (long)(nBase + 16 + lr) * CP;
    const _Float16* br2 = xb + (long)(nBase + 32 + lr) * CP;
    const _Float16* br3 = xb + (long)(nBase + 48 + lr) * CP;
    const _Float16* lA  = wtile + (long)lr * CP;

    v8f acc0 = {}, acc1 = {}, acc2 = {}, acc3 = {};
#pragma unroll
    for (int k = 0; k < KT; ++k) {
        int k0 = k * 32;
        if (k + 1 < KT) {  // near-cache prefetch of next K tile (locality=3)
            __builtin_prefetch(br0 + k0 + 32, 0, 3);
            __builtin_prefetch(br2 + k0 + 32, 0, 3);
        }
        // batch all fragment loads, then issue the 4 WMMAs back-to-back
        v16h a  = load_frag16(lA, k0, hi);   // ds_load from staged weights
        v16h b0 = load_frag16(br0, k0, hi);
        v16h b1 = load_frag16(br1, k0, hi);
        v16h b2 = load_frag16(br2, k0, hi);
        v16h b3 = load_frag16(br3, k0, hi);
        acc0 = wmma16(a, b0, acc0);
        acc1 = wmma16(a, b1, acc1);
        acc2 = wmma16(a, b2, acc2);
        acc3 = wmma16(a, b3, acc3);
    }

    // store y: C/D layout lane<16 -> (M=r,N=lane); lane>=16 -> (M=r+8,N=lane-16)
#pragma unroll
    for (int r = 0; r < 8; ++r) {
        int o = o0 + hi * 8 + r;
        float* yr = y + ((long)b * O + o) * N + nBase + lr;
        yr[0]  = acc0[r];
        yr[16] = acc1[r];
        yr[32] = acc2[r];
        yr[48] = acc3[r];
    }

    // BN stats: per-lane partials -> LDS float atomics -> global atomics
#pragma unroll
    for (int r = 0; r < 8; ++r) {
        float v0 = acc0[r], v1 = acc1[r], v2 = acc2[r], v3 = acc3[r];
        float s  = v0 + v1 + v2 + v3;
        float ss = v0 * v0 + v1 * v1 + v2 * v2 + v3 * v3;
        atomicAdd(&lsum[hi * 8 + r], s);
        atomicAdd(&lss[hi * 8 + r], ss);
    }
    __syncthreads();
    if (tid < 16) {
        atomicAdd(&chsum[o0 + tid], lsum[tid]);
        atomicAdd(&chsq[o0 + tid], lss[tid]);
    }
}

__global__ void bn_finalize_kernel(const float* __restrict__ chsum,
                                   const float* __restrict__ chsq,
                                   const float* __restrict__ gamma,
                                   const float* __restrict__ beta,
                                   float* __restrict__ A, float* __restrict__ Bb,
                                   int O, float invCount) {
    int o = blockIdx.x * blockDim.x + threadIdx.x;
    if (o >= O) return;
    float m   = chsum[o] * invCount;
    float var = chsq[o] * invCount - m * m;
    float a   = gamma[o] * rsqrtf(var + EPS);
    A[o]  = a;
    Bb[o] = beta[o] - a * m;
}

// BN+lrelu; optionally also emits next stage's (B,N,O) f16 operand (O==Cpad).
__global__ void bn_apply_lrelu_kernel(const float* __restrict__ y,
                                      const float* __restrict__ A,
                                      const float* __restrict__ Bb,
                                      float* __restrict__ out,
                                      _Float16* __restrict__ xh_next,  // nullable
                                      int B, int O, int N, int outCtot) {
    long i = (long)blockIdx.x * blockDim.x + threadIdx.x;
    long total = (long)B * O * N;
    if (i >= total) return;
    int n = (int)(i % N);
    long t = i / N;
    int o = (int)(t % O);
    int b = (int)(t / O);
    float v = A[o] * y[i] + Bb[o];
    float r = (v > 0.0f) ? v : SLOPE * v;
    out[((long)b * outCtot + o) * N + n] = r;
    if (xh_next) xh_next[((long)b * N + n) * O + o] = (_Float16)r;
}

__global__ void pool_kernel(const float* __restrict__ h5, float* __restrict__ z,
                            int B, int E, int N) {
    int be = blockIdx.x;
    int b = be / E, e = be % E;
    const float* p = h5 + ((long)b * E + e) * N;
    float mx = -3.402823466e38f, s = 0.0f;
    for (int n = threadIdx.x; n < N; n += blockDim.x) {
        float v = p[n];
        mx = fmaxf(mx, v);
        s += v;
    }
    __shared__ float smx[128];
    __shared__ float ssm[128];
    int tid = threadIdx.x;
    smx[tid] = mx; ssm[tid] = s;
    __syncthreads();
    for (int st = 64; st > 0; st >>= 1) {
        if (tid < st) {
            smx[tid] = fmaxf(smx[tid], smx[tid + st]);
            ssm[tid] += ssm[tid + st];
        }
        __syncthreads();
    }
    if (tid == 0) {
        z[(long)b * 2 * E + e]     = smx[0];
        z[(long)b * 2 * E + E + e] = ssm[0] / (float)N;
    }
}

// out[m,o] = sum_k Ah[m,k]*Bh[o,k] + bias[o] ; M=16 batch rows fit one tile.
__global__ void gemm_fc_kernel(const _Float16* __restrict__ Ah,
                               const _Float16* __restrict__ Bh,
                               const float* __restrict__ bias,
                               float* __restrict__ out, int O, int Kpad) {
    int o0 = blockIdx.x * 16;
    int lane = threadIdx.x;
    int hi = lane >> 4, lr = lane & 15;
    v8f acc = {};
    for (int k0 = 0; k0 < Kpad; k0 += 32) {
        v16h a  = load_frag16(Ah + (long)lr * Kpad, k0, hi);
        v16h bf = load_frag16(Bh + (long)(o0 + lr) * Kpad, k0, hi);
        acc = wmma16(a, bf, acc);
    }
    int o = o0 + lr;
#pragma unroll
    for (int r = 0; r < 8; ++r) {
        int m = hi * 8 + r;
        if (o < O) {
            float v = acc[r] + (bias ? bias[o] : 0.0f);
            out[(long)m * O + o] = v;
        }
    }
}

__global__ void fc_bn_lrelu_kernel(float* __restrict__ u,
                                   const float* __restrict__ gamma,
                                   const float* __restrict__ beta,
                                   int Brows, int F) {
    int f = blockIdx.x * blockDim.x + threadIdx.x;
    if (f >= F) return;
    float s = 0.0f, ss = 0.0f;
    for (int m = 0; m < Brows; ++m) {
        float v = u[(long)m * F + f];
        s += v; ss += v * v;
    }
    float mean = s / Brows;
    float var  = ss / Brows - mean * mean;
    float a  = gamma[f] * rsqrtf(var + EPS);
    float bb = beta[f] - a * mean;
    for (int m = 0; m < Brows; ++m) {
        float v = a * u[(long)m * F + f] + bb;
        u[(long)m * F + f] = (v > 0.0f) ? v : SLOPE * v;
    }
}

// ---------------------------------------------------------------------------
// Host driver
// ---------------------------------------------------------------------------
extern "C" void kernel_launch(void* const* d_in, const int* in_sizes, int n_in,
                              void* d_out, int out_size, void* d_ws, size_t ws_size,
                              hipStream_t stream) {
    (void)in_sizes; (void)n_in; (void)out_size; (void)ws_size;

    const float* x     = (const float*)d_in[0];
    const float* W1    = (const float*)d_in[1];
    const float* g1    = (const float*)d_in[2];
    const float* b1    = (const float*)d_in[3];
    const float* W2    = (const float*)d_in[4];
    const float* g2    = (const float*)d_in[5];
    const float* b2    = (const float*)d_in[6];
    const float* W3    = (const float*)d_in[7];
    const float* g3    = (const float*)d_in[8];
    const float* b3    = (const float*)d_in[9];
    const float* W4    = (const float*)d_in[10];
    const float* g4    = (const float*)d_in[11];
    const float* b4    = (const float*)d_in[12];
    const float* W5    = (const float*)d_in[13];
    const float* g5    = (const float*)d_in[14];
    const float* b5    = (const float*)d_in[15];
    const float* lin1W = (const float*)d_in[16];
    const float* g6    = (const float*)d_in[17];
    const float* b6    = (const float*)d_in[18];
    const float* lin2W = (const float*)d_in[19];
    const float* lin2b = (const float*)d_in[20];
    const float* g7    = (const float*)d_in[21];
    const float* b7    = (const float*)d_in[22];
    const float* lin3W = (const float*)d_in[23];
    const float* lin3b = (const float*)d_in[24];
    // d_in[25] = k (== 20, compiled in as KNBR)

    char* wp = (char*)d_ws;
    auto carve = [&](size_t bytes) -> void* {
        void* r = (void*)wp;
        wp += (bytes + 255) & ~(size_t)255;
        return r;
    };
    _Float16* bufXH   = (_Float16*)carve((size_t)BB * NN * 128 * 2);
    _Float16* bufAGGH = (_Float16*)carve((size_t)BB * NN * 512 * 2);
    float*    sq      = (float*)   carve((size_t)BB * NN * 4);
    int*      nbr     = (int*)     carve((size_t)BB * NN * KNBR * 4);
    float*    agg     = (float*)   carve((size_t)BB * 128 * NN * 4);
    float*    h       = (float*)   carve((size_t)BB * 512 * NN * 4);
    float*    ybuf    = (float*)   carve((size_t)BB * 1024 * NN * 4);
    _Float16* Wh      = (_Float16*)carve((size_t)512 * 2048 * 2);
    float*    chsum   = (float*)   carve((size_t)1024 * 4);
    float*    chsq    = (float*)   carve((size_t)1024 * 4);
    float*    Abuf    = (float*)   carve((size_t)1024 * 4);
    float*    Bbuf    = (float*)   carve((size_t)1024 * 4);
    float*    z       = (float*)   carve((size_t)BB * 2048 * 4);
    _Float16* zh      = (_Float16*)carve((size_t)BB * 2048 * 2);
    float*    u1      = (float*)   carve((size_t)BB * 512 * 4);
    _Float16* u1h     = (_Float16*)carve((size_t)BB * 512 * 2);
    float*    u2      = (float*)   carve((size_t)BB * 256 * 4);
    _Float16* u2h     = (_Float16*)carve((size_t)BB * 256 * 2);

    auto cdiv = [](long a, long b) -> int { return (int)((a + b - 1) / b); };

    auto conv_bn = [&](const _Float16* Bop, const float* W, const float* gamma,
                       const float* beta, int Cin, int CinPad, int Cout,
                       float* outBase, int outCtot, _Float16* xhNext) {
        cast_mat_f16<<<cdiv((long)Cout * CinPad, 256), 256, 0, stream>>>(
            W, Wh, Cout, Cin, Cout, CinPad);
        hipMemsetAsync(chsum, 0, 1024 * sizeof(float), stream);
        hipMemsetAsync(chsq, 0, 1024 * sizeof(float), stream);
        int grid = BB * (Cout / 16) * (NN / 256);
        if (CinPad == 32)
            gemm_bn_kernel<32><<<grid, 128, 0, stream>>>(Wh, Bop, ybuf, chsum, chsq, BB, Cout, NN);
        else if (CinPad == 64)
            gemm_bn_kernel<64><<<grid, 128, 0, stream>>>(Wh, Bop, ybuf, chsum, chsq, BB, Cout, NN);
        else if (CinPad == 128)
            gemm_bn_kernel<128><<<grid, 128, 0, stream>>>(Wh, Bop, ybuf, chsum, chsq, BB, Cout, NN);
        else
            gemm_bn_kernel<512><<<grid, 128, 0, stream>>>(Wh, Bop, ybuf, chsum, chsq, BB, Cout, NN);
        bn_finalize_kernel<<<cdiv(Cout, 64), 64, 0, stream>>>(
            chsum, chsq, gamma, beta, Abuf, Bbuf, Cout, 1.0f / (float)(BB * NN));
        bn_apply_lrelu_kernel<<<cdiv((long)BB * Cout * NN, 256), 256, 0, stream>>>(
            ybuf, Abuf, Bbuf, outBase, xhNext, BB, Cout, NN, outCtot);
    };

    auto edge_stage = [&](const float* src, int srcCtot, int Cin, int CinPad,
                          const float* W, const float* gamma, const float* beta,
                          int Cout, float* hslice, _Float16* xhNext) {
        // bufXH (B,N,CinPad) is produced either by the initial cast (stage 1)
        // or by the previous stage's fused bn_apply.
        sqnorm_kernel<<<cdiv((long)BB * NN, 256), 256, 0, stream>>>(
            src, sq, BB, Cin, NN, srcCtot);
        if (CinPad == 32)
            knn_topk_kernel<32><<<BB * (NN / 16), 32, 0, stream>>>(bufXH, sq, nbr, NN);
        else if (CinPad == 64)
            knn_topk_kernel<64><<<BB * (NN / 16), 32, 0, stream>>>(bufXH, sq, nbr, NN);
        else
            knn_topk_kernel<128><<<BB * (NN / 16), 32, 0, stream>>>(bufXH, sq, nbr, NN);
        gather_sum_kernel<<<BB * NN, 64, 0, stream>>>(src, nbr, agg, BB, Cin, NN, srcCtot);
        cast_bcn_to_bnc_f16<<<cdiv((long)BB * NN * CinPad, 256), 256, 0, stream>>>(
            agg, bufAGGH, BB, Cin, NN, CinPad, Cin);
        conv_bn(bufAGGH, W, gamma, beta, Cin, CinPad, Cout, hslice, 512, xhNext);
    };

    // stage 1 input cast (x is f32 (B,3,N) -> (B,N,32) f16)
    cast_bcn_to_bnc_f16<<<cdiv((long)BB * NN * 32, 256), 256, 0, stream>>>(
        x, bufXH, BB, 3, NN, 32, 3);

    // four EdgeConv blocks; each fused bn_apply also writes next knn operand
    edge_stage(x,                  3,   3,  32, W1, g1, b1,  64, h,                  bufXH);
    edge_stage(h,                512,  64,  64, W2, g2, b2,  64, h + (long)64 * NN,  bufXH);
    edge_stage(h + (long)64*NN,  512,  64,  64, W3, g3, b3, 128, h + (long)128 * NN, bufXH);
    edge_stage(h + (long)128*NN, 512, 128, 128, W4, g4, b4, 256, h + (long)256 * NN, nullptr);

    // conv5: (B,512,N) -> (B,1024,N), BN+lrelu in place on ybuf
    cast_bcn_to_bnc_f16<<<cdiv((long)BB * NN * 512, 256), 256, 0, stream>>>(
        h, bufAGGH, BB, 512, NN, 512, 512);
    conv_bn(bufAGGH, W5, g5, b5, 512, 512, 1024, ybuf, 1024, nullptr);

    // global max+mean pool -> z (B, 2048)
    pool_kernel<<<BB * 1024, 128, 0, stream>>>(ybuf, z, BB, 1024, NN);

    // fc1
    cast_mat_f16<<<cdiv((long)BB * 2048, 256), 256, 0, stream>>>(z, zh, BB, 2048, BB, 2048);
    cast_mat_f16<<<cdiv((long)512 * 2048, 256), 256, 0, stream>>>(lin1W, Wh, 512, 2048, 512, 2048);
    gemm_fc_kernel<<<512 / 16, 32, 0, stream>>>(zh, Wh, nullptr, u1, 512, 2048);
    fc_bn_lrelu_kernel<<<cdiv(512, 64), 64, 0, stream>>>(u1, g6, b6, BB, 512);

    // fc2
    cast_mat_f16<<<cdiv((long)BB * 512, 256), 256, 0, stream>>>(u1, u1h, BB, 512, BB, 512);
    cast_mat_f16<<<cdiv((long)256 * 512, 256), 256, 0, stream>>>(lin2W, Wh, 256, 512, 256, 512);
    gemm_fc_kernel<<<256 / 16, 32, 0, stream>>>(u1h, Wh, lin2b, u2, 256, 512);
    fc_bn_lrelu_kernel<<<cdiv(256, 64), 64, 0, stream>>>(u2, g7, b7, BB, 256);

    // fc3 -> d_out (16,40); O padded to 48
    cast_mat_f16<<<cdiv((long)BB * 256, 256), 256, 0, stream>>>(u2, u2h, BB, 256, BB, 256);
    cast_mat_f16<<<cdiv((long)48 * 256, 256), 256, 0, stream>>>(lin3W, Wh, 40, 256, 48, 256);
    gemm_fc_kernel<<<48 / 16, 32, 0, stream>>>(u2h, Wh, lin3b, (float*)d_out, 40, 256);
}
